// Attention_Decoder_38328288149655
// MI455X (gfx1250) — matrix operand order repaired and checked by reading, not compiled, not run on hardware
//
#include <hip/hip_runtime.h>
#include <hip/hip_bf16.h>
#include <math.h>

typedef __bf16 bf16_t;
typedef __attribute__((ext_vector_type(16))) __bf16 v16bf;
typedef __attribute__((ext_vector_type(8)))  __bf16 v8bf;
typedef __attribute__((ext_vector_type(8)))  float  v8f;

#define B_SZ     128
#define T_LEN    256
#define A_FEAT   512
#define OUT_FEAT 512
#define MID      1000
#define MID_PAD  1024
#define CAT      1024   /* OUT_FEAT + A_FEAT */
#define STEPS    32
#define GATES_N  2048   /* 4 * OUT_FEAT */
#define BT       (B_SZ * T_LEN)   /* 32768 */

// ---------------------------------------------------------------------------
// Conversion / init kernels
// ---------------------------------------------------------------------------
__global__ void k_cvt_f32_to_bf16(const float* __restrict__ src,
                                  bf16_t* __restrict__ dst, int n) {
  for (int i = blockIdx.x * blockDim.x + threadIdx.x; i < n;
       i += gridDim.x * blockDim.x)
    dst[i] = (bf16_t)src[i];
}

// dst[MID_PAD, A_FEAT] = W1[:, 0:512] in bf16, zero rows m >= MID
__global__ void k_cvt_w1a(const float* __restrict__ W1, bf16_t* __restrict__ dst) {
  const int n = MID_PAD * A_FEAT;
  for (int i = blockIdx.x * blockDim.x + threadIdx.x; i < n;
       i += gridDim.x * blockDim.x) {
    int m = i >> 9, f = i & 511;
    dst[i] = (m < MID) ? (bf16_t)W1[m * CAT + f] : (bf16_t)0.0f;
  }
}

// dst[4*OUT_FEAT, CAT] = {w_c; w_u; w_f; w_o} in bf16
__global__ void k_cvt_wg(const float* __restrict__ wc, const float* __restrict__ wu,
                         const float* __restrict__ wf, const float* __restrict__ wo,
                         bf16_t* __restrict__ dst) {
  const int per = OUT_FEAT * CAT;   // 524288
  const int n = 4 * per;
  for (int i = blockIdx.x * blockDim.x + threadIdx.x; i < n;
       i += gridDim.x * blockDim.x) {
    int g = i / per, r = i - g * per;
    const float* s = (g == 0) ? wc : (g == 1) ? wu : (g == 2) ? wf : wo;
    dst[i] = (bf16_t)s[r];
  }
}

__global__ void k_init_state(const float* __restrict__ s_prev,
                             float* __restrict__ s_state,
                             float* __restrict__ c_state) {
  const int n = B_SZ * OUT_FEAT;
  for (int i = blockIdx.x * blockDim.x + threadIdx.x; i < n;
       i += gridDim.x * blockDim.x) {
    s_state[i] = s_prev[i];
    c_state[i] = 0.0f;
  }
}

// ---------------------------------------------------------------------------
// Generic bf16 WMMA GEMM:  D[M,N] = A[M,K] * B[N,K]^T   (f32 accumulate)
// One 16x16 tile per wave, K stepped by 32 via v_wmma_f32_16x16x32_bf16.
// All operands L2-resident for our sizes -> direct global loads, no LDS.
// ---------------------------------------------------------------------------
template <bool OUT_BF16>
__global__ __launch_bounds__(256)
void k_gemm_wmma_bf16(const bf16_t* __restrict__ A,   // [M,K] row-major
                      const bf16_t* __restrict__ Bm,  // [N,K] row-major (i.e. B^T)
                      void* __restrict__ D,           // [M,N]
                      int M, int N, int K) {
  const int lane   = threadIdx.x & 31;
  const int wave   = threadIdx.x >> 5;
  const int tile   = blockIdx.x * (blockDim.x >> 5) + wave;
  const int ntiles = N >> 4;
  const int mtiles = M >> 4;
  if (tile >= mtiles * ntiles) return;     // wave-uniform exit (EXEC stays full)
  const int tm = tile / ntiles;
  const int tn = tile - tm * ntiles;

  const int h = lane >> 4;    // lane half (0/1)
  const int r = lane & 15;    // row inside tile

  const bf16_t* aRow = A  + (size_t)(tm * 16 + r) * K;
  const bf16_t* bRow = Bm + (size_t)(tn * 16 + r) * K;

  v8f acc = {};
  for (int k = 0; k < K; k += 32) {
    // A fragment: lane l holds K = h*8+0..7 (vgpr 0-3) and 16+h*8+0..7 (vgpr 4-7)
    v8bf a_lo = *(const v8bf*)(aRow + k + h * 8);
    v8bf a_hi = *(const v8bf*)(aRow + k + 16 + h * 8);
    // B fragment: lane l holds K = h*8+0..7 (vgpr 0-3) and 16+h*8+0..7 (vgpr 4-7)
    v8bf b_lo = *(const v8bf*)(bRow + k + h * 8);
    v8bf b_hi = *(const v8bf*)(bRow + k + 16 + h * 8);
    v16bf af, bf;
#pragma unroll
    for (int i = 0; i < 8; ++i) {
      af[i] = a_lo[i]; af[8 + i] = a_hi[i];
      bf[i] = b_lo[i]; bf[8 + i] = b_hi[i];
    }
    acc = __builtin_amdgcn_wmma_f32_16x16x32_bf16(
        /*neg_a=*/false, af, /*neg_b=*/false, bf,
        /*c_mod=*/(short)0, acc, /*reuse_a=*/false, /*reuse_b=*/false);
  }

  // C/D layout: lanes 0-15 -> M = i,   N = lane
  //             lanes 16-31 -> M = 8+i, N = lane-16
  const int col = tn * 16 + r;
  if (OUT_BF16) {
    bf16_t* out = (bf16_t*)D;
#pragma unroll
    for (int i = 0; i < 8; ++i)
      out[(size_t)(tm * 16 + h * 8 + i) * N + col] = (bf16_t)acc[i];
  } else {
    float* out = (float*)D;
#pragma unroll
    for (int i = 0; i < 8; ++i)
      out[(size_t)(tm * 16 + h * 8 + i) * N + col] = acc[i];
  }
}

// ---------------------------------------------------------------------------
// Per-step kernels
// ---------------------------------------------------------------------------
// Sproj[b, m] = b1[m] + dot(s_state[b, :512], W1[m, 512:1024])
__global__ __launch_bounds__(256)
void k_sproj(const float* __restrict__ W1, const float* __restrict__ b1,
             const float* __restrict__ s_state, float* __restrict__ Sproj) {
  __shared__ float sv[OUT_FEAT];
  const int tid = threadIdx.x;
  const int b = blockIdx.y;
  sv[tid]       = s_state[b * OUT_FEAT + tid];
  sv[tid + 256] = s_state[b * OUT_FEAT + tid + 256];
  __syncthreads();
  const int m = blockIdx.x * 256 + tid;
  if (m >= MID) return;
  float acc = b1[m];
  const float* wrow = W1 + (size_t)m * CAT + OUT_FEAT;
#pragma unroll 4
  for (int f = 0; f < OUT_FEAT; ++f) acc += sv[f] * wrow[f];
  Sproj[b * MID_PAD + m] = acc;
}

// logits[b,t] = relu(b2 + sum_m W2[m] * tanh(Aproj_bf16[b,t,m] + Sproj[b,m]))
__global__ __launch_bounds__(256)
void k_logits(const bf16_t* __restrict__ Aproj, const float* __restrict__ Sproj,
              const float* __restrict__ W2, const float* __restrict__ b2,
              float* __restrict__ logits) {
  __shared__ float red[256];
  const int tid = threadIdx.x;
  const int t = blockIdx.x, b = blockIdx.y;
  const bf16_t* ap = Aproj + (size_t)(b * T_LEN + t) * MID_PAD;
  const float*  sp = Sproj + (size_t)b * MID_PAD;
  float sum = 0.0f;
  for (int m = tid; m < MID; m += 256)
    sum += W2[m] * tanhf((float)ap[m] + sp[m]);
  red[tid] = sum;
  __syncthreads();
  for (int off = 128; off > 0; off >>= 1) {
    if (tid < off) red[tid] += red[tid + off];
    __syncthreads();
  }
  if (tid == 0) logits[b * T_LEN + t] = fmaxf(red[0] + b2[0], 0.0f);
}

// softmax over T, context[b,:] = alphas . a[b,:,:], and build xa_bf16 = [s, ctx]
__global__ __launch_bounds__(256)
void k_softmax_ctx(const float* __restrict__ logits, const float* __restrict__ a,
                   const float* __restrict__ s_state, bf16_t* __restrict__ xa) {
  __shared__ float red[256];
  __shared__ float alpha[T_LEN];
  const int tid = threadIdx.x;
  const int b = blockIdx.x;
  const float l = logits[b * T_LEN + tid];
  red[tid] = l;
  __syncthreads();
  for (int off = 128; off > 0; off >>= 1) {
    if (tid < off) red[tid] = fmaxf(red[tid], red[tid + off]);
    __syncthreads();
  }
  const float mx = red[0];
  __syncthreads();
  const float e = expf(l - mx);
  red[tid] = e;
  __syncthreads();
  for (int off = 128; off > 0; off >>= 1) {
    if (tid < off) red[tid] += red[tid + off];
    __syncthreads();
  }
  alpha[tid] = e / red[0];
  __syncthreads();

  // context features f = tid and tid + 256
  float c0 = 0.0f, c1 = 0.0f;
  const float* ab = a + (size_t)b * T_LEN * A_FEAT;
  for (int t = 0; t < T_LEN; ++t) {
    const float al = alpha[t];
    c0 += al * ab[t * A_FEAT + tid];
    c1 += al * ab[t * A_FEAT + tid + 256];
  }
  bf16_t* xb = xa + (size_t)b * CAT;
  xb[tid]                 = (bf16_t)s_state[b * OUT_FEAT + tid];
  xb[tid + 256]           = (bf16_t)s_state[b * OUT_FEAT + tid + 256];
  xb[OUT_FEAT + tid]       = (bf16_t)c0;
  xb[OUT_FEAT + 256 + tid] = (bf16_t)c1;
}

// LSTM pointwise: gates[b, g*512+j] (g = c,u,f,o), update c/s, emit out[b,step,j]
__global__ __launch_bounds__(256)
void k_lstm_apply(const float* __restrict__ gates,
                  const float* __restrict__ b_c, const float* __restrict__ b_u,
                  const float* __restrict__ b_f, const float* __restrict__ b_o,
                  float* __restrict__ s_state, float* __restrict__ c_state,
                  float* __restrict__ out, int step) {
  const int i = blockIdx.x * blockDim.x + threadIdx.x;   // 0 .. B*512-1
  if (i >= B_SZ * OUT_FEAT) return;
  const int b = i >> 9, j = i & 511;
  const float* g = gates + (size_t)b * GATES_N;
  const float cand = tanhf(g[0 * OUT_FEAT + j] + b_c[j]);
  const float gu = 1.0f / (1.0f + expf(-(g[1 * OUT_FEAT + j] + b_u[j])));
  const float gf = 1.0f / (1.0f + expf(-(g[2 * OUT_FEAT + j] + b_f[j])));
  const float go = 1.0f / (1.0f + expf(-(g[3 * OUT_FEAT + j] + b_o[j])));
  const float c1 = gu * cand + gf * c_state[i];
  const float a1 = go * tanhf(c1);
  c_state[i] = c1;
  s_state[i] = a1;
  out[((size_t)b * STEPS + step) * OUT_FEAT + j] = a1;
}

// ---------------------------------------------------------------------------
extern "C" void kernel_launch(void* const* d_in, const int* in_sizes, int n_in,
                              void* d_out, int out_size, void* d_ws, size_t ws_size,
                              hipStream_t stream) {
  const float* a      = (const float*)d_in[0];
  const float* s_prev = (const float*)d_in[1];
  const float* W1     = (const float*)d_in[2];
  const float* b1     = (const float*)d_in[3];
  const float* W2     = (const float*)d_in[4];
  const float* b2     = (const float*)d_in[5];
  const float* w_c    = (const float*)d_in[6];
  const float* w_u    = (const float*)d_in[7];
  const float* w_f    = (const float*)d_in[8];
  const float* w_o    = (const float*)d_in[9];
  const float* b_c    = (const float*)d_in[10];
  const float* b_u    = (const float*)d_in[11];
  const float* b_f    = (const float*)d_in[12];
  const float* b_o    = (const float*)d_in[13];
  float* out          = (float*)d_out;

  // workspace carve-up (256B aligned slabs)
  char* p = (char*)d_ws;
  size_t off = 0;
  auto take = [&](size_t bytes) {
    void* q = p + off;
    off += (bytes + 255) & ~(size_t)255;
    return q;
  };
  bf16_t* a_bf16   = (bf16_t*)take((size_t)BT * A_FEAT * 2);        // 33.5 MB
  bf16_t* W1a_bf16 = (bf16_t*)take((size_t)MID_PAD * A_FEAT * 2);   //  1.0 MB
  bf16_t* Aproj    = (bf16_t*)take((size_t)BT * MID_PAD * 2);       // 67.1 MB
  bf16_t* Wg_bf16  = (bf16_t*)take((size_t)GATES_N * CAT * 2);      //  4.2 MB
  float*  s_state  = (float*)take((size_t)B_SZ * OUT_FEAT * 4);
  float*  c_state  = (float*)take((size_t)B_SZ * OUT_FEAT * 4);
  float*  Sproj    = (float*)take((size_t)B_SZ * MID_PAD * 4);
  float*  logits   = (float*)take((size_t)B_SZ * T_LEN * 4);
  bf16_t* xa       = (bf16_t*)take((size_t)B_SZ * CAT * 2);
  float*  gates    = (float*)take((size_t)B_SZ * GATES_N * 4);
  (void)ws_size; (void)in_sizes; (void)n_in; (void)out_size;

  // one-time conversions + state init (run every call; deterministic)
  k_cvt_f32_to_bf16<<<2048, 256, 0, stream>>>(a, a_bf16, BT * A_FEAT);
  k_cvt_w1a<<<512, 256, 0, stream>>>(W1, W1a_bf16);
  k_cvt_wg<<<1024, 256, 0, stream>>>(w_c, w_u, w_f, w_o, Wg_bf16);
  k_init_state<<<256, 256, 0, stream>>>(s_prev, s_state, c_state);

  // hoisted attention projection: Aproj[BT, MID_PAD] = a_bf16 * W1a^T
  {
    const int tiles = (BT / 16) * (MID_PAD / 16);        // 131072
    k_gemm_wmma_bf16<true><<<tiles / 8, 256, 0, stream>>>(
        a_bf16, W1a_bf16, (void*)Aproj, BT, MID_PAD, A_FEAT);
  }

  for (int step = 0; step < STEPS; ++step) {
    k_sproj<<<dim3(4, B_SZ), 256, 0, stream>>>(W1, b1, s_state, Sproj);
    k_logits<<<dim3(T_LEN, B_SZ), 256, 0, stream>>>(Aproj, Sproj, W2, b2, logits);
    k_softmax_ctx<<<B_SZ, 256, 0, stream>>>(logits, a, s_state, xa);
    {
      const int tiles = (B_SZ / 16) * (GATES_N / 16);    // 1024
      k_gemm_wmma_bf16<false><<<tiles / 8, 256, 0, stream>>>(
          xa, Wg_bf16, (void*)gates, B_SZ, GATES_N, CAT);
    }
    k_lstm_apply<<<(B_SZ * OUT_FEAT + 255) / 256, 256, 0, stream>>>(
        gates, b_c, b_u, b_f, b_o, s_state, c_state, out, step);
  }
}
